// GCNLayer_88570815578698
// MI455X (gfx1250) — matrix operand order repaired and checked by reading, not compiled, not run on hardware
//
#include <hip/hip_runtime.h>

#define BN    8
#define NN    2048
#define FIN   256
#define FOUT  256
#define KSTEP 32
#define STEPS (NN / KSTEP)   // 64 k-slabs
#define LROW  40             // bf16 elems per padded LDS row (64B data + 16B pad)

typedef __bf16 bf16_t;
typedef bf16_t v16bf __attribute__((ext_vector_type(16)));
typedef float  v8f   __attribute__((ext_vector_type(8)));
typedef float  v4f   __attribute__((ext_vector_type(4)));
typedef unsigned int v8u   __attribute__((ext_vector_type(8)));
typedef unsigned int v4u   __attribute__((ext_vector_type(4)));
typedef unsigned int u32x4 __attribute__((ext_vector_type(4)));
typedef int          i32x8 __attribute__((ext_vector_type(8)));
typedef int          i32x4 __attribute__((ext_vector_type(4)));

struct BPair { u32x4 lo, hi; };

// Pack two f32 into one dword of two bf16 using the HW converter (RNE).
static __device__ __forceinline__ unsigned pk_bf16(float lo, float hi) {
  unsigned d;
  asm("v_cvt_pk_bf16_f32 %0, %1, %2" : "=v"(d) : "v"(lo), "v"(hi));
  return d;
}

// Concatenate 8+8 f32 -> v16bf (element j = j-th value).
static __device__ __forceinline__ v16bf cvt16(v8f lo, v8f hi8) {
  v8u u;
#pragma unroll
  for (int j = 0; j < 4; ++j) {
    u[j]     = pk_bf16(lo[2 * j],  lo[2 * j + 1]);
    u[4 + j] = pk_bf16(hi8[2 * j], hi8[2 * j + 1]);
  }
  return __builtin_bit_cast(v16bf, u);
}

// 32B B-fragment load (two b128s) from bf16 memory (global or LDS).
static __device__ __forceinline__ BPair load_bfrag(const bf16_t* p) {
  BPair bp;
  bp.lo = *(const u32x4*)(p);
  bp.hi = *(const u32x4*)(p + 8);
  return bp;
}

#define WMMA_BF16(A, F, ACC)                                            \
  __builtin_amdgcn_wmma_f32_16x16x32_bf16(                              \
      false, (A), false, __builtin_bit_cast(v16bf, (F)), (short)0,      \
      (ACC), false, false)

// ---------------------------------------------------------------------------
// TDM: DMA a [32 o-rows x KSTEP m] bf16 tile of Tt (row stride NN elems)
// into LDS with 4-dword padding after each 16-dword row (80B LDS pitch).
// ---------------------------------------------------------------------------
static __device__ __forceinline__ void tdm_issue(size_t gaddr, unsigned lds_addr) {
  u32x4 g0;
  g0.x = 1u;                                                  // count=1 (valid)
  g0.y = lds_addr;                                            // LDS byte addr
  g0.z = (unsigned)gaddr;                                     // global_addr[31:0]
  g0.w = (unsigned)((gaddr >> 32) & 0x01ffffffu) | (2u << 30);// [56:32] | type=2
  i32x8 g1;
  g1[0] = (int)((1u << 16)      // data_size = 1 -> 2-byte elements
              | (1u << 20)      // pad_enable
              | (3u << 22)      // pad_interval: 16 dwords
              | (3u << 25));    // pad_amount:   4 dwords
  g1[1] = (int)(((unsigned)NN & 0xffffu) << 16);              // tensor_dim0 lo16
  g1[2] = (int)((((unsigned)NN >> 16) & 0xffffu)              // tensor_dim0 hi16
              | (((unsigned)FOUT & 0xffffu) << 16));          // tensor_dim1 lo16
  g1[3] = (int)((((unsigned)FOUT >> 16) & 0xffffu)            // tensor_dim1 hi16
              | ((unsigned)KSTEP << 16));                     // tile_dim0 = 32
  g1[4] = 32;                                                 // tile_dim1=32, tile_dim2=0
  g1[5] = NN;                                                 // tensor_dim0_stride lo32
  g1[6] = 0;
  g1[7] = 0;
  i32x4 gz4 = {0, 0, 0, 0};
  i32x8 gz8 = {0, 0, 0, 0, 0, 0, 0, 0};
  __builtin_amdgcn_tensor_load_to_lds(g0, g1, gz4, gz4, gz8, 0);
}

// ---------------------------------------------------------------------------
// Kernel 0: one-shot W fp32 -> bf16 copy (keeps [FOUT][FIN] row-major).
// ---------------------------------------------------------------------------
__global__ void __launch_bounds__(256, 1)
gcn_wconv_kernel(const float* __restrict__ W, bf16_t* __restrict__ Wbf) {
  const int i = (blockIdx.x * blockDim.x + threadIdx.x) * 8;
  v8f w = *(const v8f*)(W + i);
  v4u pk;
#pragma unroll
  for (int j = 0; j < 4; ++j) pk[j] = pk_bf16(w[2 * j], w[2 * j + 1]);
  *(v4u*)(Wbf + i) = pk;
}

// ---------------------------------------------------------------------------
// Kernel 1: degree[b,n] = sum_m adj[b,n,m];  dis = rsqrt(degree) (0 if deg==0)
// ---------------------------------------------------------------------------
__global__ void __launch_bounds__(256, 1)
gcn_degree_kernel(const float* __restrict__ adj, float* __restrict__ dis) {
  const int rowId = (blockIdx.x * blockDim.x + threadIdx.x) >> 5; // 0..B*N-1
  const int lane  = threadIdx.x & 31;
  const float* rp = adj + (size_t)rowId * NN;
  float s = 0.f;
#pragma unroll
  for (int i = 0; i < NN; i += 128) {
    v4f x = *(const v4f*)(rp + i + lane * 4);
    s += x[0] + x[1] + x[2] + x[3];
  }
#pragma unroll
  for (int off = 16; off > 0; off >>= 1)
    s += __shfl_xor(s, off, 32);
  if (lane == 0)
    dis[rowId] = (s > 0.f) ? rsqrtf(s) : 0.f;
}

// ---------------------------------------------------------------------------
// Kernel 2: Tt[b][o][m] = bf16( dis[b,m] * (X[b,m,:] . W[o,:] + bias[o]) )
// 4-deep rotated B-fragment pipeline (also rotates across k-steps).
// ---------------------------------------------------------------------------
__global__ void __launch_bounds__(256, 1)
gcn_xform_kernel(const float* __restrict__ X, const bf16_t* __restrict__ Wbf,
                 const float* __restrict__ bias, const float* __restrict__ dis,
                 bf16_t* __restrict__ Tt) {
  const int b    = blockIdx.y;
  const int wave = threadIdx.x >> 5;
  const int lane = threadIdx.x & 31;
  const int r    = lane & 15;
  const int hi   = lane >> 4;
  const int m0   = blockIdx.x * 128 + wave * 16;

  const float*  Ab = X + ((size_t)b * NN + m0 + r) * FIN + hi * 16;
  const bf16_t* Wb = Wbf + (size_t)r * FIN + hi * 16;

  v8f alo = *(const v8f*)(Ab);
  v8f ahi = *(const v8f*)(Ab + 8);

  BPair f0 = load_bfrag(Wb + (size_t)0 * 16 * FIN);
  BPair f1 = load_bfrag(Wb + (size_t)1 * 16 * FIN);
  BPair f2 = load_bfrag(Wb + (size_t)2 * 16 * FIN);
  BPair f3 = load_bfrag(Wb + (size_t)3 * 16 * FIN);

  v8f acc[16] = {};
  for (int k0 = 0; k0 < FIN; k0 += 32) {
    v16bf a = cvt16(alo, ahi);
    if (k0 + 32 < FIN) {                      // prefetch next A k-slab
      alo = *(const v8f*)(Ab + k0 + 32);
      ahi = *(const v8f*)(Ab + k0 + 40);
    }
#define K2_STEP(J, FJ)                                                       \
    {                                                                        \
      acc[t + (J)] = WMMA_BF16(a, FJ, acc[t + (J)]);                         \
      const int tn = t + 4 + (J);                                            \
      if (tn < 16)                                                           \
        FJ = load_bfrag(Wb + (size_t)tn * 16 * FIN + k0);                    \
      else if (k0 + 32 < FIN)                                                \
        FJ = load_bfrag(Wb + (size_t)(tn - 16) * 16 * FIN + k0 + 32);        \
    }
#pragma unroll
    for (int t = 0; t < 16; t += 4) {
      K2_STEP(0, f0)
      K2_STEP(1, f1)
      K2_STEP(2, f2)
      K2_STEP(3, f3)
    }
#undef K2_STEP
  }

  float ds[8];
#pragma unroll
  for (int v = 0; v < 8; ++v) ds[v] = dis[b * NN + m0 + hi * 8 + v];

  bf16_t* Tbase = Tt + (size_t)b * FOUT * NN + m0 + hi * 8;
#pragma unroll
  for (int t = 0; t < 16; ++t) {
    const int o = t * 16 + r;
    const float bo = bias[o];
    v4u pk;
#pragma unroll
    for (int j = 0; j < 4; ++j)
      pk[j] = pk_bf16((acc[t][2 * j]     + bo) * ds[2 * j],
                      (acc[t][2 * j + 1] + bo) * ds[2 * j + 1]);
    *(v4u*)(Tbase + (size_t)o * NN) = pk;
  }
}

// ---------------------------------------------------------------------------
// Kernel 3: out[b,n,o] = relu( dis[b,n] * sum_m bf16(adj[b,n,m]) * Tt[b][o][m] )
// B staged through LDS by TDM (double-buffered); A streamed NT from HBM.
// 4-deep rotated LDS B-fragment pipeline within each k-slab.
// ---------------------------------------------------------------------------
__global__ void __launch_bounds__(256, 1)
gcn_aggregate_kernel(const float* __restrict__ adj, const bf16_t* __restrict__ Tt,
                     const float* __restrict__ dis, float* __restrict__ out) {
  __shared__ bf16_t smem[2][FOUT * LROW];   // 2 x 20 KB

  const int b    = blockIdx.y;
  const int tid  = threadIdx.x;
  const int wave = tid >> 5;
  const int lane = tid & 31;
  const int r    = lane & 15;
  const int hi   = lane >> 4;
  const int n0   = blockIdx.x * 128 + wave * 16;

  const float* Ab = adj + ((size_t)b * NN + n0 + r) * NN + hi * 16;

  // Each wave DMAs o-rows [wave*32, wave*32+32) of each 32-wide k-slab.
  const size_t  gslab = (size_t)(const void*)(Tt + (size_t)b * FOUT * NN
                                                 + (size_t)(wave * 32) * NN);
  const unsigned lds0 = (unsigned)(size_t)(void*)&smem[0][wave * 32 * LROW];
  const unsigned lds1 = (unsigned)(size_t)(void*)&smem[1][wave * 32 * LROW];

  tdm_issue(gslab + (size_t)0 * KSTEP * 2, lds0);
  tdm_issue(gslab + (size_t)1 * KSTEP * 2, lds1);

  v8f alo = __builtin_nontemporal_load((const v8f*)(Ab));
  v8f ahi = __builtin_nontemporal_load((const v8f*)(Ab + 8));

  v8f acc[16] = {};
  for (int i = 0; i < STEPS; ++i) {
    if (i < STEPS - 1) __builtin_amdgcn_s_wait_tensorcnt(1);
    else               __builtin_amdgcn_s_wait_tensorcnt(0);
    __syncthreads();                       // all waves' slices landed in LDS

    v16bf a = cvt16(alo, ahi);
    if (i + 1 < STEPS) {                   // prefetch next A k-slab (NT)
      const int kn = (i + 1) * KSTEP;
      alo = __builtin_nontemporal_load((const v8f*)(Ab + kn));
      ahi = __builtin_nontemporal_load((const v8f*)(Ab + kn + 8));
    }

    const bf16_t* Bs = &smem[i & 1][(size_t)r * LROW + hi * 16];
    BPair f0 = load_bfrag(Bs + (size_t)0 * 16 * LROW);
    BPair f1 = load_bfrag(Bs + (size_t)1 * 16 * LROW);
    BPair f2 = load_bfrag(Bs + (size_t)2 * 16 * LROW);
    BPair f3 = load_bfrag(Bs + (size_t)3 * 16 * LROW);
#define K3_STEP(J, FJ)                                                       \
    {                                                                        \
      acc[t + (J)] = WMMA_BF16(a, FJ, acc[t + (J)]);                         \
      const int tn = t + 4 + (J);                                            \
      if (tn < 16)                                                           \
        FJ = load_bfrag(Bs + (size_t)tn * 16 * LROW);                        \
    }
#pragma unroll
    for (int t = 0; t < 16; t += 4) {
      K3_STEP(0, f0)
      K3_STEP(1, f1)
      K3_STEP(2, f2)
      K3_STEP(3, f3)
    }
#undef K3_STEP
    __syncthreads();                       // everyone done reading this buffer
    if (i + 2 < STEPS)
      tdm_issue(gslab + (size_t)(i + 2) * KSTEP * 2, (i & 1) ? lds1 : lds0);
  }

  float ds[8];
#pragma unroll
  for (int v = 0; v < 8; ++v) ds[v] = dis[b * NN + n0 + hi * 8 + v];

#pragma unroll
  for (int t = 0; t < 16; ++t) {
    const int o = t * 16 + r;
#pragma unroll
    for (int v = 0; v < 8; ++v) {
      float val = acc[t][v] * ds[v];
      out[((size_t)b * NN + (n0 + hi * 8 + v)) * FOUT + o] = val > 0.f ? val : 0.f;
    }
  }
}

// ---------------------------------------------------------------------------
extern "C" void kernel_launch(void* const* d_in, const int* in_sizes, int n_in,
                              void* d_out, int out_size, void* d_ws, size_t ws_size,
                              hipStream_t stream) {
  const float* X    = (const float*)d_in[0]; // [B,N,FIN]
  const float* adj  = (const float*)d_in[1]; // [B,N,N]
  const float* W    = (const float*)d_in[2]; // [FOUT,FIN]
  const float* bias = (const float*)d_in[3]; // [FOUT]
  float* out = (float*)d_out;                // [B,N,FOUT]

  float*  dis = (float*)d_ws;                                    // 64 KB
  bf16_t* Tt  = (bf16_t*)((char*)d_ws + 64 * 1024);              // 8 MB
  bf16_t* Wbf = (bf16_t*)((char*)d_ws + 64 * 1024
                          + (size_t)BN * FOUT * NN * 2);         // 128 KB

  gcn_wconv_kernel<<<(FOUT * FIN) / (256 * 8), 256, 0, stream>>>(W, Wbf);
  gcn_degree_kernel<<<(BN * NN) / 8, 256, 0, stream>>>(adj, dis);
  gcn_xform_kernel<<<dim3(NN / 128, BN), 256, 0, stream>>>(X, Wbf, bias, dis, Tt);
  gcn_aggregate_kernel<<<dim3(NN / 128, BN), 256, 0, stream>>>(adj, Tt, dis, out);
}
